// GINGraphClassifier_1391569404376
// MI455X (gfx1250) — matrix operand order repaired and checked by reading, not compiled, not run on hardware
//
#include <hip/hip_runtime.h>

typedef _Float16 half4  __attribute__((ext_vector_type(4)));
typedef _Float16 half8  __attribute__((ext_vector_type(8)));
typedef _Float16 half16 __attribute__((ext_vector_type(16)));
typedef float    float8 __attribute__((ext_vector_type(8)));

#define N_NODES  50000
#define N_EDGES  800000
#define IN_CH    128
#define HID      64
#define OUT_CH   10
#define N_GRAPHS 512

// ---------------------------------------------------------------- utilities
__global__ void zero_f32(float* __restrict__ p, long n) {
  long i = (long)blockIdx.x * blockDim.x + threadIdx.x;
  if (i < n) p[i] = 0.0f;
}

// --------------------------------------------------- edge scatter (f32 src)
__global__ void scatter_add_f32(const float* __restrict__ x,
                                const int* __restrict__ src,
                                const int* __restrict__ dst,
                                float* __restrict__ agg) {
  const int CPT = IN_CH / 4;  // 32 chunks of 4 channels; one wave == one edge
  long t = (long)blockIdx.x * blockDim.x + threadIdx.x;
  if (t >= (long)N_EDGES * CPT) return;
  int e  = (int)(t / CPT);
  int c4 = (int)(t % CPT) * 4;
  int s = src[e], d = dst[e];
  const float4 v = *(const float4*)(x + (long)s * IN_CH + c4);
  float* o = agg + (long)d * IN_CH + c4;
  atomicAdd(o + 0, v.x);
  atomicAdd(o + 1, v.y);
  atomicAdd(o + 2, v.z);
  atomicAdd(o + 3, v.w);
}

// --------------------------------------------------- edge scatter (f16 src)
__global__ void scatter_add_f16(const _Float16* __restrict__ h,
                                const int* __restrict__ src,
                                const int* __restrict__ dst,
                                float* __restrict__ agg) {
  const int CPT = HID / 4;  // 16 chunks
  long t = (long)blockIdx.x * blockDim.x + threadIdx.x;
  if (t >= (long)N_EDGES * CPT) return;
  int e  = (int)(t / CPT);
  int c4 = (int)(t % CPT) * 4;
  int s = src[e], d = dst[e];
  const half4 v = *(const half4*)(h + (long)s * HID + c4);
  float* o = agg + (long)d * HID + c4;
  atomicAdd(o + 0, (float)v[0]);
  atomicAdd(o + 1, (float)v[1]);
  atomicAdd(o + 2, (float)v[2]);
  atomicAdd(o + 3, (float)v[3]);
}

// ------------------------------------- (1+eps)*x + agg  ->  f16 activations
__global__ void prep_from_f32(const float* __restrict__ x, const float* __restrict__ agg,
                              const float* __restrict__ eps, _Float16* __restrict__ out, long n) {
  long i = (long)blockIdx.x * blockDim.x + threadIdx.x;
  if (i < n) out[i] = (_Float16)((1.0f + eps[0]) * x[i] + agg[i]);
}

__global__ void prep_from_f16(const _Float16* __restrict__ x, const float* __restrict__ agg,
                              const float* __restrict__ eps, _Float16* __restrict__ out, long n) {
  long i = (long)blockIdx.x * blockDim.x + threadIdx.x;
  if (i < n) out[i] = (_Float16)((1.0f + eps[0]) * (float)x[i] + agg[i]);
}

// ------------- pack W[K][N] f32 -> per-lane B-fragment order (f16), so the
// GEMM kernel reads each 32x16 B fragment as one contiguous half16 per lane.
// Layout: out[((kt*(N/16)+nt)*32 + lane)*16 + h]
//   lane 0-15  : n = nt*16+lane,   k = kt*32 + (h&7) + (h>>3)*16       (h = 0..15)
//   lane 16-31 : n = nt*16+l&15,   k = kt*32 + 8 + (h&7) + (h>>3)*16
__global__ void pack_weights(const float* __restrict__ W, _Float16* __restrict__ out,
                             int K, int N) {
  int idx = blockIdx.x * blockDim.x + threadIdx.x;
  if (idx >= K * N) return;
  int h    = idx & 15;
  int lane = (idx >> 4) & 31;
  int rest = idx >> 9;            // 32*16 = 512 per (kt,nt)
  int ntiles = N >> 4;
  int nt = rest % ntiles;
  int kt = rest / ntiles;
  int k = kt * 32 + ((lane >> 4) << 3) + (h & 7) + ((h >> 3) << 4);
  int n = nt * 16 + (lane & 15);
  out[idx] = (_Float16)W[(long)k * N + n];
}

// --------------------------- WMMA GEMM: [M x K](f16) @ [K x 64] + bias, ReLU
// Packed B staged in LDS once per block; each wave computes TWO 16x64 output
// tiles so every B fragment feeds two v_wmma_f32_16x16x32_f16 ops.
template <int K, bool OUT_F16>
__global__ __launch_bounds__(256) void gemm_relu_wmma(
    const _Float16* __restrict__ A, const _Float16* __restrict__ Bp,
    const float* __restrict__ bias, void* __restrict__ outv, int mtiles) {
  constexpr int KT = K / 32;             // k-steps
  constexpr int BH = KT * 4 * 32 * 16;   // halves in packed B (8K or 16K B)
  __shared__ _Float16 sB[BH];

  // cooperative stage of packed weights into LDS (128-bit chunks)
  for (int i = threadIdx.x * 8; i < BH; i += blockDim.x * 8)
    *(half8*)(sB + i) = *(const half8*)(Bp + i);
  __syncthreads();

  const int lane = threadIdx.x & 31;
  const int wid  = blockIdx.x * (blockDim.x >> 5) + (threadIdx.x >> 5);
  const int t0 = wid * 2;
  const int t1 = wid * 2 + 1;
  if (t0 >= mtiles) return;
  const bool has2 = (t1 < mtiles);

  float8 acc0[4], acc1[4];
#pragma unroll
  for (int nt = 0; nt < 4; ++nt) { acc0[nt] = (float8)(0.0f); acc1[nt] = (float8)(0.0f); }

  // A fragment addressing per ISA 16-bit 16x32 layout:
  // lanes 0-15:  row m0+lane,      K halves {kb+0..7, kb+16..23}
  // lanes 16-31: row m0+(lane-16), K halves {kb+8..15, kb+24..31}
  const int rsel = lane & 15;
  const int koff = (lane >> 4) << 3;
  const _Float16* arow0 = A + (long)((t0 << 4) + rsel) * K + koff;
  const _Float16* arow1 = A + (long)(((has2 ? t1 : t0) << 4) + rsel) * K + koff;

#pragma unroll
  for (int kb = 0; kb < K; kb += 32) {
    half8 lo0 = *(const half8*)(arow0 + kb);
    half8 hi0 = *(const half8*)(arow0 + kb + 16);
    half8 lo1 = *(const half8*)(arow1 + kb);
    half8 hi1 = *(const half8*)(arow1 + kb + 16);
    half16 a0, a1;
#pragma unroll
    for (int i = 0; i < 8; ++i) {
      a0[i] = lo0[i]; a0[8 + i] = hi0[i];
      a1[i] = lo1[i]; a1[8 + i] = hi1[i];
    }
#pragma unroll
    for (int nt = 0; nt < 4; ++nt) {
      half16 b = *(const half16*)(sB + (((kb >> 5) * 4 + nt) * 32 + lane) * 16);
      acc0[nt] = __builtin_amdgcn_wmma_f32_16x16x32_f16(
          false, a0, false, b, (short)0, acc0[nt], false, false);
      acc1[nt] = __builtin_amdgcn_wmma_f32_16x16x32_f16(
          false, a1, false, b, (short)0, acc1[nt], false, false);
    }
  }

  // C layout: acc element r -> row r (lanes 0-15) / row 8+r (lanes 16-31),
  // column = nt*16 + (lane & 15)
  const int mb0 = (t0 << 4) + ((lane >> 4) << 3);
  const int mb1 = (t1 << 4) + ((lane >> 4) << 3);
#pragma unroll
  for (int nt = 0; nt < 4; ++nt) {
    const int n = (nt << 4) + rsel;
    const float bv = bias[n];
#pragma unroll
    for (int r = 0; r < 8; ++r) {
      float v0 = acc0[nt][r] + bv;
      v0 = v0 > 0.0f ? v0 : 0.0f;
      if (OUT_F16)
        ((_Float16*)outv)[(long)(mb0 + r) * HID + n] = (_Float16)v0;
      else
        ((float*)outv)[(long)(mb0 + r) * HID + n] = v0;
      if (has2) {
        float v1 = acc1[nt][r] + bv;
        v1 = v1 > 0.0f ? v1 : 0.0f;
        if (OUT_F16)
          ((_Float16*)outv)[(long)(mb1 + r) * HID + n] = (_Float16)v1;
        else
          ((float*)outv)[(long)(mb1 + r) * HID + n] = v1;
      }
    }
  }
}

// ------------------------------------------------------------ graph pooling
__global__ void pool_sum(const float* __restrict__ h, const int* __restrict__ batch,
                         float* __restrict__ pooled) {
  long t = (long)blockIdx.x * blockDim.x + threadIdx.x;
  if (t >= (long)N_NODES * HID) return;
  int i = (int)(t >> 6);
  int c = (int)(t & 63);
  atomicAdd(&pooled[(long)batch[i] * HID + c], h[t]);
}

// -------------------------------------------- classifier head + log_softmax
__global__ void classify(const float* __restrict__ pooled, const float* __restrict__ fcw,
                         const float* __restrict__ fcb, float* __restrict__ out) {
  int g = blockIdx.x * blockDim.x + threadIdx.x;
  if (g >= N_GRAPHS) return;
  float logit[OUT_CH];
#pragma unroll
  for (int o = 0; o < OUT_CH; ++o) logit[o] = fcb[o];
  for (int k = 0; k < HID; ++k) {
    float p = pooled[(long)g * HID + k];
#pragma unroll
    for (int o = 0; o < OUT_CH; ++o) logit[o] += p * fcw[k * OUT_CH + o];
  }
  float mx = logit[0];
#pragma unroll
  for (int o = 1; o < OUT_CH; ++o) mx = fmaxf(mx, logit[o]);
  float s = 0.0f;
#pragma unroll
  for (int o = 0; o < OUT_CH; ++o) s += __expf(logit[o] - mx);
  float lse = mx + __logf(s);
#pragma unroll
  for (int o = 0; o < OUT_CH; ++o) out[(long)g * OUT_CH + o] = logit[o] - lse;
}

// ---------------------------------------------------------------- launcher
extern "C" void kernel_launch(void* const* d_in, const int* in_sizes, int n_in,
                              void* d_out, int out_size, void* d_ws, size_t ws_size,
                              hipStream_t stream) {
  (void)in_sizes; (void)n_in; (void)out_size; (void)ws_size;

  const float* x    = (const float*)d_in[0];
  const int*   ei   = (const int*)d_in[1];
  const int*   bat  = (const int*)d_in[2];
  const float* eps0 = (const float*)d_in[3];
  const float* w0a  = (const float*)d_in[4];
  const float* b0a  = (const float*)d_in[5];
  const float* w0b  = (const float*)d_in[6];
  const float* b0b  = (const float*)d_in[7];
  const float* eps1 = (const float*)d_in[8];
  const float* w1a  = (const float*)d_in[9];
  const float* b1a  = (const float*)d_in[10];
  const float* w1b  = (const float*)d_in[11];
  const float* b1b  = (const float*)d_in[12];
  const float* fcw  = (const float*)d_in[13];
  const float* fcb  = (const float*)d_in[14];
  const int* src = ei;
  const int* dst = ei + N_EDGES;

  char* ws = (char*)d_ws;
  size_t off = 0;
  auto alloc = [&](size_t bytes) -> void* {
    void* p = ws + off;
    off = (off + bytes + 255) & ~(size_t)255;
    return p;
  };

  float*    agg    = (float*)   alloc((size_t)N_NODES * IN_CH * 4);  // reused for layer-1 agg
  _Float16* a16    = (_Float16*)alloc((size_t)N_NODES * IN_CH * 2);  // reused for layer-1 input
  _Float16* hA     = (_Float16*)alloc((size_t)N_NODES * HID * 2);
  _Float16* hB     = (_Float16*)alloc((size_t)N_NODES * HID * 2);
  float*    h1b    = (float*)   alloc((size_t)N_NODES * HID * 4);
  float*    pooled = (float*)   alloc((size_t)N_GRAPHS * HID * 4);
  _Float16* p0a    = (_Float16*)alloc((size_t)IN_CH * HID * 2);
  _Float16* p0b    = (_Float16*)alloc((size_t)HID * HID * 2);
  _Float16* p1a    = (_Float16*)alloc((size_t)HID * HID * 2);
  _Float16* p1b    = (_Float16*)alloc((size_t)HID * HID * 2);

  const int B = 256;
  const int mtiles = N_NODES / 16;                 // 3125 exactly
  const int gemmWaves  = (mtiles + 1) / 2;         // 2 m-tiles per wave
  const int gemmBlocks = (gemmWaves + 7) / 8;      // 8 waves per block

  // zero accumulators (every call: harness does not re-poison between replays)
  long nAgg0 = (long)N_NODES * IN_CH;
  zero_f32<<<(int)((nAgg0 + B - 1) / B), B, 0, stream>>>(agg, nAgg0);
  zero_f32<<<(N_GRAPHS * HID + B - 1) / B, B, 0, stream>>>(pooled, (long)N_GRAPHS * HID);

  // pack the four MLP weight matrices into WMMA B-fragment order
  pack_weights<<<(IN_CH * HID + B - 1) / B, B, 0, stream>>>(w0a, p0a, IN_CH, HID);
  pack_weights<<<(HID * HID + B - 1) / B, B, 0, stream>>>(w0b, p0b, HID, HID);
  pack_weights<<<(HID * HID + B - 1) / B, B, 0, stream>>>(w1a, p1a, HID, HID);
  pack_weights<<<(HID * HID + B - 1) / B, B, 0, stream>>>(w1b, p1b, HID, HID);

  // ---- layer 0
  long t0 = (long)N_EDGES * (IN_CH / 4);
  scatter_add_f32<<<(int)((t0 + B - 1) / B), B, 0, stream>>>(x, src, dst, agg);
  prep_from_f32<<<(int)((nAgg0 + B - 1) / B), B, 0, stream>>>(x, agg, eps0, a16, nAgg0);
  gemm_relu_wmma<IN_CH, true><<<gemmBlocks, 256, 0, stream>>>(a16, p0a, b0a, hA, mtiles);
  gemm_relu_wmma<HID,   true><<<gemmBlocks, 256, 0, stream>>>(hA,  p0b, b0b, hB, mtiles);

  // ---- layer 1 (agg buffer reused, zeroed after layer-0 consumption)
  long nAgg1 = (long)N_NODES * HID;
  zero_f32<<<(int)((nAgg1 + B - 1) / B), B, 0, stream>>>(agg, nAgg1);
  long t1 = (long)N_EDGES * (HID / 4);
  scatter_add_f16<<<(int)((t1 + B - 1) / B), B, 0, stream>>>(hB, src, dst, agg);
  prep_from_f16<<<(int)((nAgg1 + B - 1) / B), B, 0, stream>>>(hB, agg, eps1, a16, nAgg1);
  gemm_relu_wmma<HID, true ><<<gemmBlocks, 256, 0, stream>>>(a16, p1a, b1a, hA, mtiles);
  gemm_relu_wmma<HID, false><<<gemmBlocks, 256, 0, stream>>>(hA,  p1b, b1b, h1b, mtiles);

  // ---- pooling + classifier
  long tp = (long)N_NODES * HID;
  pool_sum<<<(int)((tp + B - 1) / B), B, 0, stream>>>(h1b, bat, pooled);
  classify<<<(N_GRAPHS + B - 1) / B, B, 0, stream>>>(pooled, fcw, fcb, (float*)d_out);
}
